// DGCNN_7516192768970
// MI455X (gfx1250) — compile-verified, matrix operand order checked by reading, not run on hardware
//
#include <hip/hip_runtime.h>
#include <hip/hip_bf16.h>
#include <math.h>

// Problem constants (match reference)
constexpr int NN    = 262144;   // nodes
constexpr int NG    = 2048;     // graphs
constexpr int NPER  = 128;      // nodes per graph
constexpr int NE    = 1048576;  // edges
constexpr int DIMF  = 128;
constexpr int EMB   = 64;
constexpr int TOT   = 257;
constexpr int KSEL  = 79;
constexpr int PL    = 39;       // pooled length
constexpr int C2L   = 35;       // conv2 output length
constexpr int DIN   = 1120;     // 35*32
constexpr int DHID  = 128;

typedef __bf16 bf16;
typedef __attribute__((ext_vector_type(8)))  __bf16 v8bf;
typedef __attribute__((ext_vector_type(16))) __bf16 v16bf;
typedef __attribute__((ext_vector_type(8)))  float  v8f;

// ---------------- degree / normalization ----------------
__global__ void k_deg_init(float* deg) {
  int v = blockIdx.x * blockDim.x + threadIdx.x;
  if (v < NN) deg[v] = 1.0f;                      // self loop
}
__global__ void k_deg_count(const int* __restrict__ src, float* deg) {
  int e = blockIdx.x * blockDim.x + threadIdx.x;
  if (e < NE) atomicAdd(&deg[src[e]], 1.0f);
}
__global__ void k_deg_inv(float* deg) {
  int v = blockIdx.x * blockDim.x + threadIdx.x;
  if (v < NN) deg[v] = 1.0f / deg[v];
}

// ---------------- fp32 -> bf16 ----------------
__global__ void k_f2bf(const float* __restrict__ in, bf16* __restrict__ out, long n) {
  long t = (long)blockIdx.x * blockDim.x + threadIdx.x;
  if (t < n) out[t] = (bf16)in[t];
}

// ---------------- weight packing to WMMA B-operand layout ----------------
// layout index: ((kt*ntiles + nt)*32 + lane)*16 + e
// lane<16 holds col N=lane, lane>=16 holds col N=lane-16 with k-offset +8.
// element e: e<8 -> klocal = base + e ; e>=8 -> klocal = 16 + base + (e-8)
__global__ void k_pack_w(const float* __restrict__ w, bf16* __restrict__ wp,
                         int Kin, int Nout) {
  int total = Kin * Nout;
  int t = blockIdx.x * blockDim.x + threadIdx.x;
  if (t >= total) return;
  int ntiles = Nout >> 4;
  int e    = t & 15;
  int lane = (t >> 4) & 31;
  int nt   = (t >> 9) % ntiles;
  int kt   = (t >> 9) / ntiles;
  int col  = nt * 16 + (lane & 15);
  int base = (lane < 16) ? 0 : 8;
  int klocal = (e < 8) ? (base + e) : (16 + base + (e - 8));
  int k = kt * 32 + klocal;
  wp[t] = (bf16)w[k * Nout + col];
}

// ---------------- WMMA GEMM: out[NN,Nout] = act[NN,Kin](bf16) @ W + bias ----------------
// grid.x = NN/16 (m-tiles), block = 128 (4 waves, one n-tile of 16 cols each)
__global__ void k_gemm_wmma(const bf16* __restrict__ act, const bf16* __restrict__ wp,
                            const float* __restrict__ bias, float* __restrict__ out,
                            int Kin, int Nout) {
  const int mtile  = blockIdx.x;
  const int wave   = threadIdx.x >> 5;
  const int lane   = threadIdx.x & 31;
  const int ntiles = Nout >> 4;              // 4 for Nout=64 -> all waves live, EXEC full
  const int row    = mtile * 16 + (lane & 15);
  const int kshift = (lane < 16) ? 0 : 8;
  const int ktiles = Kin >> 5;

  v8f c = {};
  for (int kt = 0; kt < ktiles; ++kt) {
    union { v16bf v; v8bf h[2]; } a, b;
    const bf16* ap = act + (size_t)row * Kin + kt * 32 + kshift;
    a.h[0] = *(const v8bf*)(ap);
    a.h[1] = *(const v8bf*)(ap + 16);
    const bf16* bp = wp + (((size_t)(kt * ntiles + wave) * 32) + lane) * 16;
    b.h[0] = *(const v8bf*)(bp);
    b.h[1] = *(const v8bf*)(bp + 8);
    c = __builtin_amdgcn_wmma_f32_16x16x32_bf16(false, a.v, false, b.v,
                                                (short)0, c, false, false);
  }
  const int col   = wave * 16 + (lane & 15);
  const int rbase = mtile * 16 + ((lane < 16) ? 0 : 8);
  const float bv  = bias[col];
#pragma unroll
  for (int r = 0; r < 8; ++r)
    out[(size_t)(rbase + r) * Nout + col] = c[r] + bv;
}

// ---------------- message passing (64-wide layers) ----------------
__global__ void k_copy4(const float* __restrict__ h, float* __restrict__ agg, long n4) {
  long t = (long)blockIdx.x * blockDim.x + threadIdx.x;
  if (t < n4) ((float4*)agg)[t] = ((const float4*)h)[t];   // agg = h (self loop)
}
__global__ void k_scatter64(const int* __restrict__ src, const int* __restrict__ dst,
                            const float* __restrict__ h, float* __restrict__ agg) {
  long t = (long)blockIdx.x * blockDim.x + threadIdx.x;
  if (t >= (long)NE * 16) return;
  int e  = (int)(t >> 4);
  int f4 = (int)(t & 15) << 2;
  float4 v = *(const float4*)(h + (size_t)src[e] * EMB + f4);
  float* o = agg + (size_t)dst[e] * EMB + f4;
  atomicAdd(o + 0, v.x); atomicAdd(o + 1, v.y);
  atomicAdd(o + 2, v.z); atomicAdd(o + 3, v.w);
}
__global__ void k_finish64(const float* __restrict__ agg, const float* __restrict__ invdeg,
                           float* __restrict__ xc, bf16* __restrict__ actnext, int layer) {
  long t = (long)blockIdx.x * blockDim.x + threadIdx.x;
  if (t >= (long)NN * EMB) return;
  int v = (int)(t >> 6), f = (int)(t & 63);
  float val = tanhf(invdeg[v] * agg[t]);
  xc[(size_t)v * TOT + layer * EMB + f] = val;
  actnext[t] = (bf16)val;
}

// ---------------- conv5 (Kin=64, Nout=1) ----------------
__global__ void k_conv5(const bf16* __restrict__ act, const float* __restrict__ w,
                        const float* __restrict__ b, float* __restrict__ h1,
                        float* __restrict__ agg1) {
  int v = blockIdx.x * blockDim.x + threadIdx.x;
  if (v >= NN) return;
  float s = b[0];
  const bf16* a = act + (size_t)v * EMB;
#pragma unroll 8
  for (int k = 0; k < EMB; ++k) s += (float)a[k] * w[k];
  h1[v] = s; agg1[v] = s;                       // self loop seeded
}
__global__ void k_scatter1(const int* __restrict__ src, const int* __restrict__ dst,
                           const float* __restrict__ h1, float* __restrict__ agg1) {
  int e = blockIdx.x * blockDim.x + threadIdx.x;
  if (e < NE) atomicAdd(&agg1[dst[e]], h1[src[e]]);
}
__global__ void k_finish1(const float* __restrict__ agg1, const float* __restrict__ invdeg,
                          float* __restrict__ xc) {
  int v = blockIdx.x * blockDim.x + threadIdx.x;
  if (v < NN) xc[(size_t)v * TOT + 256] = tanhf(invdeg[v] * agg1[v]);
}

// ---------------- sort pool: top-KSEL per graph by channel 256 desc ----------------
__global__ void k_sortpool(const float* __restrict__ xc, int* __restrict__ order) {
  __shared__ float keys[NPER];
  int g = blockIdx.x, i = threadIdx.x;
  float ki = xc[((size_t)g * NPER + i) * TOT + 256];
  keys[i] = ki;
  __syncthreads();
  int rank = 0;
  for (int j = 0; j < NPER; ++j) {
    float kj = keys[j];
    rank += (kj > ki) || (kj == ki && j < i);   // stable argsort(-key)
  }
  if (rank < KSEL) order[g * KSEL + rank] = i;
}

// ---------------- c1: per selected node dot with 16 filters of len 257, relu ----------------
__global__ void k_c1(const float* __restrict__ xc, const int* __restrict__ order,
                     const float* __restrict__ w, const float* __restrict__ bias,
                     float* __restrict__ c1out) {
  long t = (long)blockIdx.x * blockDim.x + threadIdx.x;
  if (t >= (long)NG * 16 * KSEL) return;
  int k = (int)(t % KSEL);
  int c = (int)((t / KSEL) % 16);
  int g = (int)(t / (KSEL * 16));
  int node = g * NPER + order[g * KSEL + k];
  const float* row = xc + (size_t)node * TOT;
  const float* f   = w + c * TOT;
  float s = bias[c];
  for (int d = 0; d < TOT; ++d) s += row[d] * f[d];
  c1out[t] = fmaxf(s, 0.0f);
}

// ---------------- maxpool 2x2: 79 -> 39 ----------------
__global__ void k_pool(const float* __restrict__ c1out, float* __restrict__ pool) {
  long t = (long)blockIdx.x * blockDim.x + threadIdx.x;
  if (t >= (long)NG * 16 * PL) return;
  int tp = (int)(t % PL);
  long rest = t / PL;
  const float* p = c1out + rest * KSEL + 2 * tp;
  pool[t] = fmaxf(p[0], p[1]);
}

// ---------------- conv1d 16->32, k=5, relu, write [B,1120] flat ----------------
__global__ void k_c2(const float* __restrict__ pool, const float* __restrict__ w,
                     const float* __restrict__ bias, float* __restrict__ c2out) {
  long t = (long)blockIdx.x * blockDim.x + threadIdx.x;
  if (t >= (long)NG * 32 * C2L) return;
  int tt = (int)(t % C2L);
  int o  = (int)((t / C2L) % 32);
  int g  = (int)(t / (C2L * 32));
  float s = bias[o];
  for (int i = 0; i < 16; ++i) {
    const float* p = pool + ((size_t)g * 16 + i) * PL + tt;
    const float* f = w + (o * 16 + i) * 5;
#pragma unroll
    for (int tau = 0; tau < 5; ++tau) s += p[tau] * f[tau];
  }
  c2out[(size_t)g * DIN + o * C2L + tt] = fmaxf(s, 0.0f);
}

// ---------------- dense 1120 -> 128, relu ----------------
__global__ void k_d1(const float* __restrict__ flat, const float* __restrict__ w,
                     const float* __restrict__ bias, float* __restrict__ hid) {
  long t = (long)blockIdx.x * blockDim.x + threadIdx.x;
  if (t >= (long)NG * DHID) return;
  int j = (int)(t % DHID);
  int g = (int)(t / DHID);
  const float* x = flat + (size_t)g * DIN;
  float s = bias[j];
  for (int i = 0; i < DIN; ++i) s += x[i] * w[(size_t)i * DHID + j];
  hid[t] = fmaxf(s, 0.0f);
}

// ---------------- dense 128 -> 1 ----------------
__global__ void k_d2(const float* __restrict__ hid, const float* __restrict__ w,
                     const float* __restrict__ bias, float* __restrict__ out) {
  int g = blockIdx.x * blockDim.x + threadIdx.x;
  if (g >= NG) return;
  const float* x = hid + (size_t)g * DHID;
  float s = bias[0];
#pragma unroll 8
  for (int j = 0; j < DHID; ++j) s += x[j] * w[j];
  out[g] = s;
}

static inline int blks(long n, int b) { return (int)((n + b - 1) / b); }

extern "C" void kernel_launch(void* const* d_in, const int* in_sizes, int n_in,
                              void* d_out, int out_size, void* d_ws, size_t ws_size,
                              hipStream_t stream) {
  // inputs (setup_inputs dict order, params flattened in insertion order)
  const float* x    = (const float*)d_in[0];
  const int*   ei   = (const int*)d_in[1];
  const int*   src  = ei;
  const int*   dst  = ei + NE;
  // d_in[2] = batch (graphs are contiguous 128-node blocks; unused)
  const float* cw[5] = {(const float*)d_in[3], (const float*)d_in[5], (const float*)d_in[7],
                        (const float*)d_in[9], (const float*)d_in[11]};
  const float* cb[5] = {(const float*)d_in[4], (const float*)d_in[6], (const float*)d_in[8],
                        (const float*)d_in[10], (const float*)d_in[12]};
  const float* c1w = (const float*)d_in[13]; const float* c1b = (const float*)d_in[14];
  const float* c2w = (const float*)d_in[15]; const float* c2b = (const float*)d_in[16];
  const float* d1w = (const float*)d_in[17]; const float* d1b = (const float*)d_in[18];
  const float* d2w = (const float*)d_in[19]; const float* d2b = (const float*)d_in[20];
  float* out = (float*)d_out;

  // workspace carve-up (256B aligned)
  char* p = (char*)d_ws; size_t off = 0;
  auto carve = [&](size_t bytes) -> char* {
    char* r = p + off; off = (off + bytes + 255) & ~(size_t)255; return r;
  };
  float* invdeg = (float*)carve((size_t)NN * 4);
  bf16*  actbf  = (bf16*) carve((size_t)NN * DIMF * 2);
  float* hbuf   = (float*)carve((size_t)NN * EMB * 4);
  float* aggbuf = (float*)carve((size_t)NN * EMB * 4);
  float* xc     = (float*)carve((size_t)NN * TOT * 4);
  int*   order  = (int*)  carve((size_t)NG * KSEL * 4);
  float* c1out  = (float*)carve((size_t)NG * 16 * KSEL * 4);
  float* pool   = (float*)carve((size_t)NG * 16 * PL * 4);
  float* c2out  = (float*)carve((size_t)NG * DIN * 4);
  float* d1h    = (float*)carve((size_t)NG * DHID * 4);
  bf16*  wpack  = (bf16*) carve((size_t)DIMF * EMB * 2);
  (void)ws_size; (void)in_sizes; (void)n_in; (void)out_size;

  const int T = 256;

  // 1/deg with self loops (norm depends only on dst -> folded to per-node scale)
  k_deg_init<<<blks(NN, T), T, 0, stream>>>(invdeg);
  k_deg_count<<<blks(NE, T), T, 0, stream>>>(src, invdeg);
  k_deg_inv<<<blks(NN, T), T, 0, stream>>>(invdeg);

  // layer-0 input in bf16
  k_f2bf<<<blks((long)NN * DIMF, T), T, 0, stream>>>(x, actbf, (long)NN * DIMF);

  // 4 GCN layers with EMB=64 outputs (WMMA GEMM + scatter + norm + tanh)
  for (int l = 0; l < 4; ++l) {
    int Kin = (l == 0) ? DIMF : EMB;
    k_pack_w<<<blks(Kin * EMB, T), T, 0, stream>>>(cw[l], wpack, Kin, EMB);
    k_gemm_wmma<<<NN / 16, 128, 0, stream>>>(actbf, wpack, cb[l], hbuf, Kin, EMB);
    k_copy4<<<blks((long)NN * EMB / 4, T), T, 0, stream>>>(hbuf, aggbuf, (long)NN * EMB / 4);
    k_scatter64<<<blks((long)NE * 16, T), T, 0, stream>>>(src, dst, hbuf, aggbuf);
    k_finish64<<<blks((long)NN * EMB, T), T, 0, stream>>>(aggbuf, invdeg, xc, actbf, l);
  }

  // layer 4: 64 -> 1 (scalar), writes channel 256 of xc
  k_conv5<<<blks(NN, T), T, 0, stream>>>(actbf, cw[4], cb[4], hbuf, aggbuf);
  k_scatter1<<<blks(NE, T), T, 0, stream>>>(src, dst, hbuf, aggbuf);
  k_finish1<<<blks(NN, T), T, 0, stream>>>(aggbuf, invdeg, xc);

  // sort pool (top-79 indices per graph)
  k_sortpool<<<NG, NPER, 0, stream>>>(xc, order);

  // readout head
  k_c1<<<blks((long)NG * 16 * KSEL, T), T, 0, stream>>>(xc, order, c1w, c1b, c1out);
  k_pool<<<blks((long)NG * 16 * PL, T), T, 0, stream>>>(c1out, pool);
  k_c2<<<blks((long)NG * 32 * C2L, T), T, 0, stream>>>(pool, c2w, c2b, c2out);
  k_d1<<<blks((long)NG * DHID, T), T, 0, stream>>>(c2out, d1w, d1b, d1h);
  k_d2<<<blks(NG, T), T, 0, stream>>>(d1h, d2w, d2b, out);
}
